// EdgeBlock_71425306132749
// MI455X (gfx1250) — matrix operand order repaired and checked by reading, not compiled, not run on hardware
//
#include <hip/hip_runtime.h>

typedef __attribute__((ext_vector_type(16))) _Float16 v16h;
typedef __attribute__((ext_vector_type(8)))  float    v8f;
typedef __attribute__((ext_vector_type(4)))  float    f32x4;

#define D_EDGE 64
#define D_VERT 64
#define D_IN   192
#define D_HID  128
#define D_OUT  64

#define TILE_M 128
#define X_STRIDE 200   // 192 + 8 pad (f16 units); 100 words, 100%64=36 -> conflict-free
#define W1_STRIDE 200
#define H_STRIDE 136   // 128 + 8 pad; 68 words, 68%64=4 -> conflict-free
#define W2_STRIDE 136

union Frag { v16h v; f32x4 q[2]; };

// Load A fragment (16x32 f16) per ISA 7.12.2: elem i -> K = k0 + (i<8?i:i+8) + 8*(lane>=16)
__device__ __forceinline__ v16h load_a_frag(const _Float16* base) {
    Frag f;
    f.q[0] = *(const f32x4*)(base);        // elems 0..7
    f.q[1] = *(const f32x4*)(base + 16);   // elems 8..15
    return f.v;
}
// Load B fragment (32x16 f16): elem i -> K = k0 + (i<8?i:i+8) + 8*(lane>=16), N = lane&15
__device__ __forceinline__ v16h load_b_frag(const _Float16* base) {
    Frag f;
    f.q[0] = *(const f32x4*)(base);
    f.q[1] = *(const f32x4*)(base + 16);
    return f.v;
}

__global__ __launch_bounds__(256)
void edgeblock_wmma_kernel(const float* __restrict__ edata,
                           const float* __restrict__ vfeat,
                           const float* __restrict__ W1,
                           const float* __restrict__ b1,
                           const float* __restrict__ W2,
                           const float* __restrict__ b2,
                           const int*   __restrict__ senders,
                           const int*   __restrict__ receivers,
                           float* __restrict__ out,
                           int n_edges) {
    __shared__ __align__(16) _Float16 xs [TILE_M * X_STRIDE];   // 51200 B
    __shared__ __align__(16) _Float16 w1t[D_HID  * W1_STRIDE];  // 51200 B
    __shared__ __align__(16) _Float16 hs [TILE_M * H_STRIDE];   // 34816 B
    __shared__ __align__(16) _Float16 w2t[D_OUT  * W2_STRIDE];  // 17408 B
    __shared__ int ridx[TILE_M];
    __shared__ int sidx[TILE_M];

    const int tid  = threadIdx.x;
    const int lane = tid & 31;
    const int wave = tid >> 5;
    const int ln   = lane & 15;
    const int half = lane >> 4;           // 0 or 1
    const long long e0 = (long long)blockIdx.x * TILE_M;

    // ---- stage gather indices ----
    if (tid < TILE_M) {
        long long e = e0 + tid;
        if (e >= n_edges) e = n_edges - 1;
        ridx[tid] = receivers[e];
        sidx[tid] = senders[e];
    }

    // ---- stage W1^T and W2^T to LDS as f16 (coalesced reads) ----
    #pragma unroll 4
    for (int i = tid; i < D_IN * D_HID; i += 256) {
        int k = i >> 7, n = i & 127;      // W1 is [K=192][N=128] row-major
        w1t[n * W1_STRIDE + k] = (_Float16)W1[i];
    }
    #pragma unroll 4
    for (int i = tid; i < D_HID * D_OUT; i += 256) {
        int k = i >> 6, n = i & 63;       // W2 is [K=128][N=64] row-major
        w2t[n * W2_STRIDE + k] = (_Float16)W2[i];
    }
    __syncthreads();

    // ---- gather x tile [128 x 192] (f32 -> f16), float4 granularity ----
    #pragma unroll 4
    for (int c = tid; c < TILE_M * 48; c += 256) {
        int row = c / 48;
        int seg = c - row * 48;           // 48 float4 per row
        long long e = e0 + row;
        if (e >= n_edges) e = n_edges - 1;
        const float* src;
        if (seg < 16)       src = edata + e * (long long)D_EDGE + seg * 4;
        else if (seg < 32)  src = vfeat + (long long)ridx[row] * D_VERT + (seg - 16) * 4;
        else                src = vfeat + (long long)sidx[row] * D_VERT + (seg - 32) * 4;
        f32x4 v = *(const f32x4*)src;
        union { _Float16 h[4]; unsigned long long u; } pk;
        pk.h[0] = (_Float16)v.x; pk.h[1] = (_Float16)v.y;
        pk.h[2] = (_Float16)v.z; pk.h[3] = (_Float16)v.w;
        *(unsigned long long*)&xs[row * X_STRIDE + seg * 4] = pk.u;
    }
    __syncthreads();

    // ---- matmul1: h = relu(x @ W1 + b1), each wave owns N-tile n0 = wave*16 ----
    {
        const int n0 = wave * 16;
        v16h bw[6];
        #pragma unroll
        for (int ks = 0; ks < 6; ++ks)
            bw[ks] = load_b_frag(&w1t[(n0 + ln) * W1_STRIDE + ks * 32 + half * 8]);
        const float bias = b1[n0 + ln];
        for (int mt = 0; mt < 8; ++mt) {
            v8f acc;
            #pragma unroll
            for (int r = 0; r < 8; ++r) acc[r] = bias;
            #pragma unroll
            for (int ks = 0; ks < 6; ++ks) {
                v16h a = load_a_frag(&xs[(mt * 16 + ln) * X_STRIDE + ks * 32 + half * 8]);
                acc = __builtin_amdgcn_wmma_f32_16x16x32_f16(
                        false, a, false, bw[ks], (short)0, acc, false, false);
            }
            #pragma unroll
            for (int r = 0; r < 8; ++r) {
                float v = acc[r] > 0.0f ? acc[r] : 0.0f;     // fused ReLU
                hs[(mt * 16 + r + half * 8) * H_STRIDE + n0 + ln] = (_Float16)v;
            }
        }
    }
    __syncthreads();

    // ---- matmul2: out = h @ W2 + b2; wave -> (ntile = wave&3, mtiles (wave>>2)*4..+3) ----
    {
        const int n0    = (wave & 3) * 16;
        const int mbase = (wave >> 2) * 4;
        v16h bw[4];
        #pragma unroll
        for (int ks = 0; ks < 4; ++ks)
            bw[ks] = load_b_frag(&w2t[(n0 + ln) * W2_STRIDE + ks * 32 + half * 8]);
        const float bias = b2[n0 + ln];
        for (int j = 0; j < 4; ++j) {
            const int mt = mbase + j;
            v8f acc;
            #pragma unroll
            for (int r = 0; r < 8; ++r) acc[r] = bias;
            #pragma unroll
            for (int ks = 0; ks < 4; ++ks) {
                v16h a = load_a_frag(&hs[(mt * 16 + ln) * H_STRIDE + ks * 32 + half * 8]);
                acc = __builtin_amdgcn_wmma_f32_16x16x32_f16(
                        false, a, false, bw[ks], (short)0, acc, false, false);
            }
            #pragma unroll
            for (int r = 0; r < 8; ++r) {
                long long e = e0 + mt * 16 + r + half * 8;
                if (e < n_edges)
                    out[e * (long long)D_OUT + n0 + ln] = acc[r];
            }
        }
    }
}

extern "C" void kernel_launch(void* const* d_in, const int* in_sizes, int n_in,
                              void* d_out, int out_size, void* d_ws, size_t ws_size,
                              hipStream_t stream) {
    const float* edata     = (const float*)d_in[0];
    const float* vfeat     = (const float*)d_in[1];
    const float* W1        = (const float*)d_in[2];
    const float* b1        = (const float*)d_in[3];
    const float* W2        = (const float*)d_in[4];
    const float* b2        = (const float*)d_in[5];
    const int*   senders   = (const int*)d_in[6];
    const int*   receivers = (const int*)d_in[7];
    float*       out       = (float*)d_out;

    const int n_edges = in_sizes[6];                 // flat count of senders
    const int nblocks = (n_edges + TILE_M - 1) / TILE_M;

    edgeblock_wmma_kernel<<<nblocks, 256, 0, stream>>>(
        edata, vfeat, W1, b1, W2, b2, senders, receivers, out, n_edges);
}
